// CoreferenceResolver_14877766713684
// MI455X (gfx1250) — matrix operand order repaired and checked by reading, not compiled, not run on hardware
//
#include <hip/hip_runtime.h>
#include <hip/hip_bf16.h>
#include <stdint.h>

// ---------------------------------------------------------------------------
// CoreferenceResolver fused pipeline for MI455X (gfx1250, wave32, WMMA).
//
// feats@w1 = sim*w1[0] + emb_i@W1A + emb_j@W1B  ->  precompute EA/EB with
// bf16 WMMA GEMMs, fuse the 768->512->256->2 MLP per 64-pair tile in LDS.
// All WMMA operands that come from global memory are pre-packed into the
// native CDNA5 fragment layout so hot loops issue only global_load_b128.
// ---------------------------------------------------------------------------

namespace {

constexpr int kB     = 16;
constexpr int kL     = 2048;
constexpr int kH     = 768;
constexpr int kNE    = 64;
constexpr int kSpan  = 4;
constexpr int kPairs = (kNE * (kNE - 1)) / 2;        // 2016
constexpr int kTiles = (kPairs + 63) / 64;           // 32 pair-tiles per batch
constexpr int kH2    = 512;
constexpr int kH3    = 256;

typedef __attribute__((ext_vector_type(16))) __bf16        v16bf;
typedef __attribute__((ext_vector_type(8)))  float         v8f;
typedef __attribute__((ext_vector_type(4)))  unsigned int  v4u;

union Frag16 { v16bf v; unsigned int u[8]; };

__device__ __forceinline__ unsigned short f32_to_bf16(float f) {
  unsigned u = __float_as_uint(f);
  u += 0x7fffu + ((u >> 16) & 1u);                   // round-to-nearest-even
  return (unsigned short)(u >> 16);
}
__device__ __forceinline__ float bf16_to_f32(unsigned short h) {
  return __uint_as_float(((unsigned)h) << 16);
}

// Fragment pre-packed in register layout: lane l owns dwords [l*8, l*8+8).
// p must already include the +lane*8 offset. Two b128 loads per fragment.
__device__ __forceinline__ v16bf load_frag_packed(const unsigned int* p) {
  Frag16 f;
  *(v4u*)&f.u[0] = *(const v4u*)p;
  *(v4u*)&f.u[4] = *(const v4u*)(p + 4);
  return f.v;
}

// A fragment (16x32 bf16, MxK) gathered from a row-major LDS tile, ld elems.
// Layout per CDNA5 ISA 7.12.2 (16-bit A 16x32).
__device__ __forceinline__ v16bf load_a_frag_lds(const unsigned short* base, int ld) {
  int lane  = threadIdx.x & 31;
  int row   = lane & 15;
  int khalf = (lane & 16) ? 8 : 0;
  const unsigned short* rp = base + (size_t)row * ld;
  Frag16 f;
#pragma unroll
  for (int j = 0; j < 8; ++j) {
    int k0 = khalf + ((j & 4) ? 16 : 0) + 2 * (j & 3);
    f.u[j] = *(const unsigned int*)(rp + k0);        // k0 even -> dword aligned
  }
  return f.v;
}

// C/D f32 16x16: lane<16 -> rows 0..7, lane>=16 -> rows 8..15; col = lane&15.
__device__ __forceinline__ void store_c_frag(float* base, int ld, v8f c) {
  int lane  = threadIdx.x & 31;
  int col   = lane & 15;
  int rbase = (lane & 16) ? 8 : 0;
#pragma unroll
  for (int r = 0; r < 8; ++r) base[(size_t)(rbase + r) * ld + col] = c[r];
}

// bias + relu + bf16 pack into an LDS activation tile.
__device__ __forceinline__ void store_act_frag(unsigned short* dst, int ld,
                                               v8f c, const float* __restrict__ bias,
                                               int colbase) {
  int lane  = threadIdx.x & 31;
  int col   = colbase + (lane & 15);
  int rbase = (lane & 16) ? 8 : 0;
  float bv  = bias[col];
#pragma unroll
  for (int r = 0; r < 8; ++r) {
    float v = c[r] + bv;
    v = v > 0.0f ? v : 0.0f;
    dst[(size_t)(rbase + r) * ld + col] = f32_to_bf16(v);
  }
}

} // namespace

// ---------------------------------------------------------------------------
// K0a: pack f32 weight W[K,N] into B-fragment layout (bf16), fragment-major:
//      frag(nt,kt) at dword offset ((nt*Ks + kt) * 256), dword = lane*8 + j.
// ---------------------------------------------------------------------------
__global__ void k_pack_b(const float* __restrict__ src,
                         unsigned int* __restrict__ dst, int K, int N) {
  int g = blockIdx.x * blockDim.x + threadIdx.x;
  int Ks = K >> 5, Nt = N >> 4;
  int total = Ks * Nt * 256;
  if (g >= total) return;
  int frag = g >> 8, within = g & 255;
  int lane = within >> 3, j = within & 7;
  int nt = frag / Ks, kt = frag - nt * Ks;
  int col = nt * 16 + (lane & 15);
  int k0  = kt * 32 + ((lane & 16) ? 8 : 0) + ((j & 4) ? 16 : 0) + 2 * (j & 3);
  unsigned lo = f32_to_bf16(src[(size_t)k0 * N + col]);
  unsigned hi = f32_to_bf16(src[(size_t)(k0 + 1) * N + col]);
  dst[g] = lo | (hi << 16);
}

// ---------------------------------------------------------------------------
// K0b: pack f32 matrix A[M,K] into A-fragment layout (bf16), fragment-major:
//      frag(mt,kt) at dword offset ((mt*Ks + kt) * 256), dword = lane*8 + j.
// ---------------------------------------------------------------------------
__global__ void k_pack_a(const float* __restrict__ src,
                         unsigned int* __restrict__ dst, int M, int K) {
  int g = blockIdx.x * blockDim.x + threadIdx.x;
  int Ks = K >> 5, Mt = M >> 4;
  int total = Mt * Ks * 256;
  if (g >= total) return;
  int frag = g >> 8, within = g & 255;
  int lane = within >> 3, j = within & 7;
  int mt = frag / Ks, kt = frag - mt * Ks;
  int row = mt * 16 + (lane & 15);
  int k0  = kt * 32 + ((lane & 16) ? 8 : 0) + ((j & 4) ? 16 : 0) + 2 * (j & 3);
  unsigned lo = f32_to_bf16(src[(size_t)row * K + k0]);
  unsigned hi = f32_to_bf16(src[(size_t)row * K + k0 + 1]);
  dst[g] = lo | (hi << 16);
}

// ---------------------------------------------------------------------------
// K1: per-entity span max-pool. One block per (b,e).
// ---------------------------------------------------------------------------
__global__ void k_span_max(const float* __restrict__ x,
                           const int* __restrict__ starts,
                           float* __restrict__ emb_f32) {
  int be = blockIdx.x;                       // b*64 + e
  int b  = be >> 6;
  int s0 = starts[be];
  const float* base = x + ((size_t)b * kL + s0) * kH;
  for (int c = threadIdx.x; c < kH; c += blockDim.x) {
    float m = base[c];
#pragma unroll
    for (int s = 1; s < kSpan; ++s) {
      float v = base[(size_t)s * kH + c];
      m = v > m ? v : m;
    }
    emb_f32[(size_t)be * kH + c] = m;
  }
}

// ---------------------------------------------------------------------------
// K2: per-batch cosine matrix, unbiased std, sim = triu((cos-thr)/(std+eps)).
// One block per batch; normalized rows live in LDS (64x768 f32 = 192KB,
// within the 320KB/WGP CDNA5 LDS budget).
// ---------------------------------------------------------------------------
__global__ void k_cos_sim(const float* __restrict__ emb,
                          const float* __restrict__ thr_p,
                          float* __restrict__ cosbuf,
                          float* __restrict__ simbuf) {
  extern __shared__ unsigned char smem[];
  float* nsh = (float*)smem;                           // 64*768 f32
  float* red = (float*)(smem + (size_t)kNE * kH * 4);  // 512 f32
  int b = blockIdx.x;
  int t = threadIdx.x;
  int wave = t >> 5, lane = t & 31;

  // normalize rows (8 waves x 8 rows each)
  for (int rr = 0; rr < 8; ++rr) {
    int r = wave * 8 + rr;
    const float* row = emb + ((size_t)b * kNE + r) * kH;
    float s = 0.0f;
    for (int c = lane; c < kH; c += 32) { float v = row[c]; s += v * v; }
#pragma unroll
    for (int m = 16; m; m >>= 1) s += __shfl_xor(s, m, 32);
    float inv = 1.0f / fmaxf(sqrtf(s), 1e-8f);
    for (int c = lane; c < kH; c += 32) nsh[r * kH + c] = row[c] * inv;
  }
  __syncthreads();

  // cos(i,j) + running sum / sumsq for std
  float lsum = 0.0f, lsq = 0.0f;
  for (int idx = t; idx < kNE * kNE; idx += 256) {
    int i = idx >> 6, j = idx & 63;
    const float* ni = nsh + i * kH;
    const float* nj = nsh + j * kH;
    float d = 0.0f;
    for (int k = 0; k < kH; k += 4)
      d += ni[k] * nj[k] + ni[k + 1] * nj[k + 1]
         + ni[k + 2] * nj[k + 2] + ni[k + 3] * nj[k + 3];
    cosbuf[(size_t)b * kNE * kNE + idx] = d;
    lsum += d; lsq += d * d;
  }
  red[t] = lsum; red[256 + t] = lsq;
  __syncthreads();
  for (int s = 128; s > 0; s >>= 1) {
    if (t < s) { red[t] += red[t + s]; red[256 + t] += red[256 + t + s]; }
    __syncthreads();
  }
  const float N = (float)(kNE * kNE);
  float mean  = red[0] / N;
  float var   = (red[256] - N * mean * mean) / (N - 1.0f);   // ddof=1
  float scale = 1.0f / (sqrtf(var) + 1e-5f);
  float thr   = thr_p[0];

  for (int idx = t; idx < kNE * kNE; idx += 256) {
    int i = idx >> 6, j = idx & 63;
    float v = (j > i)
            ? (cosbuf[(size_t)b * kNE * kNE + idx] - thr) * scale : 0.0f;
    simbuf[(size_t)b * kNE * kNE + idx] = v;
  }
}

// ---------------------------------------------------------------------------
// K3: C[M,N] = A @ W (both pre-packed bf16 fragments), f32 WMMA accumulate.
// Block tile 64x64; 8 waves, each wave a 16x32 region (2 WMMAs per K step).
// All fragment traffic is contiguous global_load_b128.
// ---------------------------------------------------------------------------
__global__ void k_gemm_packed(const unsigned int* __restrict__ Ap,
                              const unsigned int* __restrict__ Bp,
                              float* __restrict__ C, int N, int K) {
  int wave = threadIdx.x >> 5;
  int lane = threadIdx.x & 31;
  int wm = wave & 3, wn = wave >> 2;
  int row0 = blockIdx.x * 64 + wm * 16;
  int col0 = blockIdx.y * 64 + wn * 32;
  int Ks = K >> 5;
  const unsigned int* ap  = Ap + (size_t)(row0 >> 4) * Ks * 256 + lane * 8;
  const unsigned int* bp0 = Bp + (size_t)(col0 >> 4) * Ks * 256 + lane * 8;
  const unsigned int* bp1 = bp0 + (size_t)Ks * 256;
  v8f acc0 = {}, acc1 = {};
  for (int kt = 0; kt < Ks; ++kt) {
    v16bf a  = load_frag_packed(ap);
    v16bf b0 = load_frag_packed(bp0);
    v16bf b1 = load_frag_packed(bp1);
    acc0 = __builtin_amdgcn_wmma_f32_16x16x32_bf16(false, a, false, b0,
                                                   (short)0, acc0, false, false);
    acc1 = __builtin_amdgcn_wmma_f32_16x16x32_bf16(false, a, false, b1,
                                                   (short)0, acc1, false, false);
    ap += 256; bp0 += 256; bp1 += 256;
  }
  store_c_frag(C + (size_t)row0 * N + col0, N, acc0);
  store_c_frag(C + (size_t)row0 * N + col0 + 16, N, acc1);
}

// ---------------------------------------------------------------------------
// K4: fused pair MLP. Grid (32 pair-tiles, 16 batches), 256 threads.
// LDS: h1 64x768 bf16 | h2 64x512 bf16 | pair meta. h3 reuses h1 space.
// Weights arrive pre-packed in B-fragment layout (b128 loads only).
// ---------------------------------------------------------------------------
__global__ void k_pair_mlp(const float* __restrict__ EA,
                           const float* __restrict__ EB,
                           const float* __restrict__ simbuf,
                           const float* __restrict__ w1,   // row 0 = sim weight
                           const float* __restrict__ b1,
                           const unsigned int* __restrict__ w2p,
                           const float* __restrict__ b2,
                           const unsigned int* __restrict__ w3p,
                           const float* __restrict__ b3,
                           const float* __restrict__ w4,
                           const float* __restrict__ b4,
                           float* __restrict__ out) {
  extern __shared__ unsigned char smem[];
  unsigned short* h1 = (unsigned short*)smem;              // 64*768
  unsigned short* h2 = h1 + 64 * kH;                       // 64*512
  unsigned char* meta = smem + (size_t)(64 * kH + 64 * kH2) * 2;
  int*   si   = (int*)meta;
  int*   sj   = si + 64;
  float* ssim = (float*)(sj + 64);
  unsigned short* h3 = h1;                                 // reuse h1 region

  int tile = blockIdx.x;
  int b    = blockIdx.y;
  int t    = threadIdx.x;
  int wave = t >> 5;
  int lane = t & 31;

  // pair index decode (np.triu_indices(64, k=1) row-major order)
  if (t < 64) {
    int p = tile * 64 + t;
    if (p >= kPairs) p = kPairs - 1;                       // clamp; store masked later
    int i = 0, rem = p;
    while (rem >= 63 - i) { rem -= 63 - i; ++i; }
    int j = i + 1 + rem;
    si[t] = i; sj[t] = j;
    ssim[t] = simbuf[(size_t)b * kNE * kNE + i * kNE + j];
  }
  __syncthreads();

  // layer 1: h1 = relu(EA[i] + EB[j] + sim*w1row0 + b1), float4 vectorized
  for (int q = t; q < 64 * (kH / 4); q += 256) {
    int r  = q / (kH / 4);
    int c4 = (q - r * (kH / 4)) * 4;
    const float4 ea = *(const float4*)(EA + ((size_t)b * kNE + si[r]) * kH + c4);
    const float4 eb = *(const float4*)(EB + ((size_t)b * kNE + sj[r]) * kH + c4);
    const float4 wr = *(const float4*)(w1 + c4);
    const float4 bb = *(const float4*)(b1 + c4);
    float sv = ssim[r];
    float v0 = ea.x + eb.x + sv * wr.x + bb.x;
    float v1 = ea.y + eb.y + sv * wr.y + bb.y;
    float v2 = ea.z + eb.z + sv * wr.z + bb.z;
    float v3 = ea.w + eb.w + sv * wr.w + bb.w;
    v0 = v0 > 0.0f ? v0 : 0.0f;  v1 = v1 > 0.0f ? v1 : 0.0f;
    v2 = v2 > 0.0f ? v2 : 0.0f;  v3 = v3 > 0.0f ? v3 : 0.0f;
    unsigned d0 = (unsigned)f32_to_bf16(v0) | ((unsigned)f32_to_bf16(v1) << 16);
    unsigned d1 = (unsigned)f32_to_bf16(v2) | ((unsigned)f32_to_bf16(v3) << 16);
    unsigned int* dst = (unsigned int*)(h1 + r * kH + c4);
    dst[0] = d0; dst[1] = d1;
  }
  __syncthreads();

  // layer 2: h2[64,512] = relu(h1 @ w2 + b2); 64 units of 16x32, 8 per wave
  {
    const int Ks = kH >> 5;                                // 24
    for (int u = wave; u < 64; u += 8) {
      int mt = u & 3;
      int col0 = (u >> 2) * 32;
      const unsigned int* bp0 = w2p + (size_t)(col0 >> 4) * Ks * 256 + lane * 8;
      const unsigned int* bp1 = bp0 + (size_t)Ks * 256;
      v8f acc0 = {}, acc1 = {};
      for (int kt = 0; kt < Ks; ++kt) {
        v16bf a  = load_a_frag_lds(h1 + mt * 16 * kH + kt * 32, kH);
        v16bf b0 = load_frag_packed(bp0);
        v16bf b1v = load_frag_packed(bp1);
        acc0 = __builtin_amdgcn_wmma_f32_16x16x32_bf16(false, a, false, b0,
                                                       (short)0, acc0, false, false);
        acc1 = __builtin_amdgcn_wmma_f32_16x16x32_bf16(false, a, false, b1v,
                                                       (short)0, acc1, false, false);
        bp0 += 256; bp1 += 256;
      }
      store_act_frag(h2 + mt * 16 * kH2, kH2, acc0, b2, col0);
      store_act_frag(h2 + mt * 16 * kH2, kH2, acc1, b2, col0 + 16);
    }
  }
  __syncthreads();

  // layer 3: h3[64,256] = relu(h2 @ w3 + b3); 32 units of 16x32, 4 per wave
  {
    const int Ks = kH2 >> 5;                               // 16
    for (int u = wave; u < 32; u += 8) {
      int mt = u & 3;
      int col0 = (u >> 2) * 32;
      const unsigned int* bp0 = w3p + (size_t)(col0 >> 4) * Ks * 256 + lane * 8;
      const unsigned int* bp1 = bp0 + (size_t)Ks * 256;
      v8f acc0 = {}, acc1 = {};
      for (int kt = 0; kt < Ks; ++kt) {
        v16bf a  = load_a_frag_lds(h2 + mt * 16 * kH2 + kt * 32, kH2);
        v16bf b0 = load_frag_packed(bp0);
        v16bf b1v = load_frag_packed(bp1);
        acc0 = __builtin_amdgcn_wmma_f32_16x16x32_bf16(false, a, false, b0,
                                                       (short)0, acc0, false, false);
        acc1 = __builtin_amdgcn_wmma_f32_16x16x32_bf16(false, a, false, b1v,
                                                       (short)0, acc1, false, false);
        bp0 += 256; bp1 += 256;
      }
      store_act_frag(h3 + mt * 16 * kH3, kH3, acc0, b3, col0);
      store_act_frag(h3 + mt * 16 * kH3, kH3, acc1, b3, col0 + 16);
    }
  }
  __syncthreads();

  // layer 4: logits[64,2] = h3 @ w4 + b4 (tiny VALU tail), masked store
  if (t < 128) {
    int r = t >> 1, c = t & 1;
    int p = tile * 64 + r;
    if (p < kPairs) {
      float acc = b4[c];
      const unsigned int* h3row = (const unsigned int*)(h3 + r * kH3);
      for (int k2 = 0; k2 < kH3 / 2; ++k2) {
        unsigned u = h3row[k2];
        acc += bf16_to_f32((unsigned short)(u & 0xffff)) * w4[(2 * k2) * 2 + c];
        acc += bf16_to_f32((unsigned short)(u >> 16))    * w4[(2 * k2 + 1) * 2 + c];
      }
      out[((size_t)b * kPairs + p) * 2 + c] = acc;
    }
  }
}

// ---------------------------------------------------------------------------
// host launcher
// ---------------------------------------------------------------------------
extern "C" void kernel_launch(void* const* d_in, const int* in_sizes, int n_in,
                              void* d_out, int out_size, void* d_ws, size_t ws_size,
                              hipStream_t stream) {
  (void)in_sizes; (void)n_in; (void)out_size; (void)ws_size;
  const float* x    = (const float*)d_in[0];
  const float* thr  = (const float*)d_in[1];
  const float* w1   = (const float*)d_in[2];
  const float* b1   = (const float*)d_in[3];
  const float* w2   = (const float*)d_in[4];
  const float* b2   = (const float*)d_in[5];
  const float* w3   = (const float*)d_in[6];
  const float* b3   = (const float*)d_in[7];
  const float* w4   = (const float*)d_in[8];
  const float* b4   = (const float*)d_in[9];
  const int* starts = (const int*)d_in[10];
  float* out = (float*)d_out;

  char* ws = (char*)d_ws;
  size_t off = 0;
  auto alloc = [&](size_t bytes) -> char* {
    char* p = ws + off;
    off = (off + bytes + 255) & ~(size_t)255;
    return p;
  };
  const int M = kB * kNE;                                   // 1024
  float*        emb_f32 = (float*)alloc((size_t)M * kH * 4);
  float*        cosbuf  = (float*)alloc((size_t)kB * kNE * kNE * 4);
  float*        simbuf  = (float*)alloc((size_t)kB * kNE * kNE * 4);
  float*        EA      = (float*)alloc((size_t)M * kH * 4);
  float*        EB      = (float*)alloc((size_t)M * kH * 4);
  unsigned int* embAp   = (unsigned int*)alloc((size_t)(M / 16) * (kH / 32) * 1024);
  unsigned int* w1aP    = (unsigned int*)alloc((size_t)(kH / 16) * (kH / 32) * 1024);
  unsigned int* w1bP    = (unsigned int*)alloc((size_t)(kH / 16) * (kH / 32) * 1024);
  unsigned int* w2P     = (unsigned int*)alloc((size_t)(kH2 / 16) * (kH / 32) * 1024);
  unsigned int* w3P     = (unsigned int*)alloc((size_t)(kH3 / 16) * (kH2 / 32) * 1024);

  // weight fragment packing (W1A = w1 rows 1..768, W1B = rows 769..1536)
  {
    int tot = (kH / 32) * (kH / 16) * 256;
    k_pack_b<<<(tot + 255) / 256, 256, 0, stream>>>(w1 + kH, w1aP, kH, kH);
    k_pack_b<<<(tot + 255) / 256, 256, 0, stream>>>(w1 + (size_t)(kH + 1) * kH,
                                                    w1bP, kH, kH);
  }
  {
    int tot = (kH / 32) * (kH2 / 16) * 256;
    k_pack_b<<<(tot + 255) / 256, 256, 0, stream>>>(w2, w2P, kH, kH2);
  }
  {
    int tot = (kH2 / 32) * (kH3 / 16) * 256;
    k_pack_b<<<(tot + 255) / 256, 256, 0, stream>>>(w3, w3P, kH2, kH3);
  }

  // entity span max-pool, then pack emb into A-fragment layout
  k_span_max<<<kB * kNE, 256, 0, stream>>>(x, starts, emb_f32);
  {
    int tot = (M / 16) * (kH / 32) * 256;
    k_pack_a<<<(tot + 255) / 256, 256, 0, stream>>>(emb_f32, embAp, M, kH);
  }

  // per-batch cosine similarity + normalized sim matrix
  size_t lds_cos = (size_t)kNE * kH * 4 + 512 * 4;
  k_cos_sim<<<kB, 256, lds_cos, stream>>>(emb_f32, thr, cosbuf, simbuf);

  // EA = emb @ W1A, EB = emb @ W1B  (M=1024, N=768, K=768)
  dim3 ggrid(M / 64, kH / 64);
  k_gemm_packed<<<ggrid, 256, 0, stream>>>(embAp, w1aP, EA, kH, kH);
  k_gemm_packed<<<ggrid, 256, 0, stream>>>(embAp, w1bP, EB, kH, kH);

  // fused pair MLP
  size_t lds_mlp = (size_t)(64 * kH + 64 * kH2) * 2 + 64 * 12;
  dim3 mgrid(kTiles, kB);
  k_pair_mlp<<<mgrid, 256, lds_mlp, stream>>>(EA, EB, simbuf, w1, b1,
                                              w2P, b2, w3P, b3, w4, b4, out);
}
